// RelPositionMultiHeadAttention_old_618475291223
// MI455X (gfx1250) — matrix operand order repaired and checked by reading, not compiled, not run on hardware
//
#include <hip/hip_runtime.h>
#include <hip/hip_bf16.h>

typedef __attribute__((ext_vector_type(16))) _Float16 v16h;
typedef __attribute__((ext_vector_type(8)))  float    v8f;

#define QLEN   1024
#define KLEN   2048
#define DMODEL 512
#define NHEAD  8
#define DK     64
#define BATCH  2

// ---------------------------------------------------------------------------
// WMMA helpers (v_wmma_f32_16x16x32_f16, wave32)
// ---------------------------------------------------------------------------
__device__ __forceinline__ v8f wmma16(v16h a, v16h b, v8f c) {
  return __builtin_amdgcn_wmma_f32_16x16x32_f16(
      /*neg_a=*/false, a, /*neg_b=*/false, b,
      /*c_mod=*/(short)0, c, /*reuse_a=*/false, /*reuse_b=*/false);
}

// A fragment: 16x32 f16 tile (row-major in LDS, given stride).
// Lane L holds row M = L%16; lanes 0-15 hold K={0..7,16..23}, lanes 16-31
// hold K={8..15,24..31} (ISA 7.12.2 16-bit A layout).
__device__ __forceinline__ v16h load_a_frag(const _Float16* t, int stride) {
  const int lane = threadIdx.x & 31;
  const _Float16* row = t + (lane & 15) * stride;
  const int hb = (lane >> 4) * 8;
  v16h a;
#pragma unroll
  for (int r = 0; r < 4; ++r) {
    a[2 * r]         = row[hb + 2 * r];
    a[2 * r + 1]     = row[hb + 2 * r + 1];
    a[8 + 2 * r]     = row[16 + hb + 2 * r];
    a[8 + 2 * r + 1] = row[16 + hb + 2 * r + 1];
  }
  return a;
}

// B fragment: 32(K) x 16(N), sourced from an LDS tile stored [n][k]
// (i.e. the transpose of B). Lane L holds column N = L%16; lanes 0-15 hold
// K={0..7,16..23}, lanes 16-31 hold K={8..15,24..31}.
__device__ __forceinline__ v16h load_b_frag(const _Float16* t, int stride) {
  const int lane = threadIdx.x & 31;
  const _Float16* col = t + (lane & 15) * stride + (lane >> 4) * 8;
  v16h b;
#pragma unroll
  for (int k = 0; k < 16; ++k) b[k] = col[k < 8 ? k : k + 8];
  return b;
}

// Async global->LDS copy of 16 bytes per lane (GLOBAL_LOAD_ASYNC_TO_LDS_B128,
// ASYNCcnt). LDS destination is the low 32 bits of the generic address
// (aperture base occupies the high bits per the flat-addressing rules).
__device__ __forceinline__ void async_copy_b128(const _Float16* gsrc, void* lds_dst) {
  unsigned dst = (unsigned)(uintptr_t)lds_dst;
  asm volatile("global_load_async_to_lds_b128 %0, %1, off"
               :: "v"(dst), "v"((unsigned long long)(uintptr_t)gsrc)
               : "memory");
}

__device__ __forceinline__ void async_drain() {
  asm volatile("s_wait_asynccnt 0x0" ::: "memory");
}

// ---------------------------------------------------------------------------
// Kernel 1/4: C[M,N] = A[M,K] @ W[N,K]^T (+bias[n]); A rows remappable so the
// same kernel computes q = key[:, -qlen:] @ Wq^T. f32->f16 conversion happens
// while staging into LDS (hence VGPR-path staging, not async byte copy).
// Grid: (N/64, M/64); 256 threads; each of 8 waves owns two 16x16 C tiles.
// ---------------------------------------------------------------------------
template <typename AT, typename OT>
__global__ __launch_bounds__(256) void gemm_wt_kernel(
    const AT* __restrict__ A, int lda, int rpb_bits, int rpb_in,
    const float* __restrict__ W, int ldw, const float* __restrict__ bias,
    OT* __restrict__ C, int ldc, int K) {
  __shared__ _Float16 As[64][34];
  __shared__ _Float16 Ws[64][34];
  const int m0 = blockIdx.y * 64, n0 = blockIdx.x * 64;
  const int wv = threadIdx.x >> 5, lane = threadIdx.x & 31;
  const int nsub = wv & 3, msub = (wv >> 2) * 2;
  const int rmask = (1 << rpb_bits) - 1;
  const int radd  = rpb_in - (1 << rpb_bits);
  v8f acc0 = {}, acc1 = {};

  for (int kk = 0; kk < K; kk += 32) {
    for (int t = threadIdx.x; t < 64 * 32; t += 256) {
      int r = t >> 5, c = t & 31;
      int m = m0 + r;
      int src = ((m >> rpb_bits) * rpb_in) + radd + (m & rmask);
      As[r][c] = (_Float16)(float)A[(size_t)src * lda + kk + c];
      Ws[r][c] = (_Float16)W[(size_t)(n0 + r) * ldw + kk + c];
    }
    __syncthreads();
    v16h bf = load_b_frag(&Ws[nsub * 16][0], 34);
    v16h a0 = load_a_frag(&As[msub * 16][0], 34);
    v16h a1 = load_a_frag(&As[(msub + 1) * 16][0], 34);
    acc0 = wmma16(a0, bf, acc0);
    acc1 = wmma16(a1, bf, acc1);
    __syncthreads();
  }

  const int ncol = n0 + nsub * 16 + (lane & 15);
  const float bv = bias ? bias[ncol] : 0.f;
  const int hi = lane >> 4;
#pragma unroll
  for (int r = 0; r < 8; ++r) {
    int mA = m0 + msub * 16 + r + 8 * hi;  // C layout: VGPR r -> M = r + 8*hi
    C[(size_t)mA * ldc + ncol]        = (OT)(acc0[r] + bv);
    C[(size_t)(mA + 16) * ldc + ncol] = (OT)(acc1[r] + bv);
  }
}

// ---------------------------------------------------------------------------
// Kernel 2: pre-shift BD scores. S[b,h,i,r] = (q[b,i,h]+v_bias[h]) . p[r,h]
// Pure f16 GEMM per (b,h): [1024 x 64] @ [64 x 2048]; tiles staged with
// async global->LDS B128 copies. Grid (KLEN/64, QLEN/64, B*H).
// ---------------------------------------------------------------------------
__global__ __launch_bounds__(256) void bd_scores_kernel(
    const _Float16* __restrict__ QV, const _Float16* __restrict__ P,
    _Float16* __restrict__ S) {
  __shared__ __align__(16) _Float16 Aq[64][72];
  __shared__ __align__(16) _Float16 Bp[64][72];
  const int z = blockIdx.z, b = z >> 3, h = z & 7;
  const int m0 = blockIdx.y * 64, n0 = blockIdx.x * 64;

  for (int t = threadIdx.x; t < 64 * 8; t += 256) {
    int r = t >> 3, c = (t & 7) * 8;  // 8-half (16 B) chunks
    async_copy_b128(QV + (size_t)(b * QLEN + m0 + r) * DMODEL + h * DK + c,
                    &Aq[r][c]);
    async_copy_b128(P + (size_t)(n0 + r) * DMODEL + h * DK + c, &Bp[r][c]);
  }
  async_drain();
  __syncthreads();

  const int wv = threadIdx.x >> 5, lane = threadIdx.x & 31;
  const int nsub = wv & 3, msub = (wv >> 2) * 2;
  v8f acc0 = {}, acc1 = {};
#pragma unroll
  for (int kk = 0; kk < 64; kk += 32) {
    v16h bf = load_b_frag(&Bp[nsub * 16][kk], 72);
    v16h a0 = load_a_frag(&Aq[msub * 16][kk], 72);
    v16h a1 = load_a_frag(&Aq[(msub + 1) * 16][kk], 72);
    acc0 = wmma16(a0, bf, acc0);
    acc1 = wmma16(a1, bf, acc1);
  }

  const int ncol = n0 + nsub * 16 + (lane & 15);
  const int hi = lane >> 4;
#pragma unroll
  for (int r = 0; r < 8; ++r) {
    int mA = m0 + msub * 16 + r + 8 * hi;
    S[((size_t)z * QLEN + mA) * KLEN + ncol]      = (_Float16)acc0[r];
    S[((size_t)z * QLEN + mA + 16) * KLEN + ncol] = (_Float16)acc1[r];
  }
}

// ---------------------------------------------------------------------------
// Kernel 3: fused flash attention per (b,h): streams 32-wide k-tiles.
//   e = (Qu.K^T + shift(SBD)) / sqrt(dk), mask, online softmax, O += P~ . V.
// rel_shift on a contiguous [i][r] buffer is pure flat-index arithmetic:
//   rr = j + qlen-1 - i;  rr<klen -> flat[i*klen+rr]; rr==klen -> 0 (pad col);
//   rr>klen (wrap into next row, faithful to reference) -> flat[i*klen+rr-1].
// Q and K tiles staged with async global->LDS; V needs a transpose so it
// keeps the VGPR staging path. Grid: (QLEN/128, B*H); 8 waves x 16 q-rows.
// ---------------------------------------------------------------------------
__global__ __launch_bounds__(256) void attn_kernel(
    const _Float16* __restrict__ QU, const _Float16* __restrict__ K16,
    const _Float16* __restrict__ V16, const _Float16* __restrict__ SBD,
    const int* __restrict__ mask, _Float16* __restrict__ CV) {
  __shared__ __align__(16) _Float16 Qs[128][72];  // q-tile, per head slice
  __shared__ __align__(16) _Float16 Ks[32][72];   // K tile [j][d]
  __shared__ _Float16 Vts[64][34];                // V tile transposed [d][j]
  __shared__ _Float16 Ps[8][16][34];              // per-wave prob staging

  const int z = blockIdx.y, b = z >> 3, h = z & 7;
  const int i0 = blockIdx.x * 128;
  const int wv = threadIdx.x >> 5, lane = threadIdx.x & 31;
  const int hi = lane >> 4, nl = lane & 15;
  const int iw = i0 + wv * 16;

  // async-stage the whole 128-row q-tile (16 KiB), overlapped with O init
  for (int t = threadIdx.x; t < 128 * 8; t += 256) {
    int r = t >> 3, c = (t & 7) * 8;
    async_copy_b128(QU + (size_t)(b * QLEN + i0 + r) * DMODEL + h * DK + c,
                    &Qs[r][c]);
  }

  v8f oacc[4] = {};
  float mrow[8], lrow[8];
#pragma unroll
  for (int r = 0; r < 8; ++r) { mrow[r] = -3.0e38f; lrow[r] = 0.f; }

  for (int j0 = 0; j0 < KLEN; j0 += 32) {
    // K tile: one async B128 per thread (32 rows x 64 halves)
    {
      int r = threadIdx.x >> 3, c = (threadIdx.x & 7) * 8;
      async_copy_b128(K16 + (size_t)(b * KLEN + j0 + r) * DMODEL + h * DK + c,
                      &Ks[r][c]);
    }
    // V tile transposed: VGPR path
    for (int t = threadIdx.x; t < 32 * 64; t += 256) {
      int r = t >> 6, c = t & 63;
      Vts[c][r] = V16[(size_t)(b * KLEN + j0 + r) * DMODEL + h * DK + c];
    }
    async_drain();
    __syncthreads();

    // AC: two 16x16 score tiles per wave, dk=64 in two K=32 WMMA steps
    v8f s0 = {}, s1 = {};
#pragma unroll
    for (int kk = 0; kk < 64; kk += 32) {
      v16h aq = load_a_frag(&Qs[wv * 16][kk], 72);
      s0 = wmma16(aq, load_b_frag(&Ks[0][kk], 72), s0);
      s1 = wmma16(aq, load_b_frag(&Ks[16][kk], 72), s1);
    }

    // add rel-shifted BD (L2-resident gather), mask, scale by 1/sqrt(64)
    float e0[8], e1[8];
#pragma unroll
    for (int r = 0; r < 8; ++r) {
      const int ig = iw + r + 8 * hi;
      const size_t srow = ((size_t)z * QLEN + ig) * KLEN;
      const size_t mri = ((size_t)b * QLEN + ig) * KLEN;
#pragma unroll
      for (int ts = 0; ts < 2; ++ts) {
        int jg = j0 + ts * 16 + nl;
        int rr = jg + (QLEN - 1) - ig;
        float bd = (rr == KLEN)
                       ? 0.f
                       : (float)SBD[srow + (size_t)(rr < KLEN ? rr : rr - 1)];
        int mk = mask[mri + jg];
        float sv = (ts == 0) ? s0[r] : s1[r];
        float ev = mk ? (sv + bd) * 0.125f : -3.0e38f;
        if (ts == 0) e0[r] = ev; else e1[r] = ev;
      }
    }

    // online softmax; rows live in the C-layout VGPR dim, columns in lanes,
    // so reductions stay inside 16-lane halves (masks 1,2,4,8).
#pragma unroll
    for (int r = 0; r < 8; ++r) {
      float t = fmaxf(e0[r], e1[r]);
      t = fmaxf(t, __shfl_xor(t, 1));
      t = fmaxf(t, __shfl_xor(t, 2));
      t = fmaxf(t, __shfl_xor(t, 4));
      t = fmaxf(t, __shfl_xor(t, 8));
      float mn = fmaxf(mrow[r], t);
      float alpha = __expf(mrow[r] - mn);
      mrow[r] = mn;
      float p0 = __expf(e0[r] - mn);
      float p1 = __expf(e1[r] - mn);
      Ps[wv][r + 8 * hi][nl]      = (_Float16)p0;
      Ps[wv][r + 8 * hi][16 + nl] = (_Float16)p1;
      float ps = p0 + p1;
      ps += __shfl_xor(ps, 1);
      ps += __shfl_xor(ps, 2);
      ps += __shfl_xor(ps, 4);
      ps += __shfl_xor(ps, 8);
      lrow[r] = lrow[r] * alpha + ps;
#pragma unroll
      for (int c = 0; c < 4; ++c) oacc[c][r] *= alpha;
    }

    // drain own LDS stores of Ps before re-reading as an A fragment
    asm volatile("s_wait_dscnt 0" ::: "memory");

    v16h ap = load_a_frag(&Ps[wv][0][0], 34);
#pragma unroll
    for (int c = 0; c < 4; ++c)
      oacc[c] = wmma16(ap, load_b_frag(&Vts[c * 16][0], 34), oacc[c]);

    __syncthreads();
  }

#pragma unroll
  for (int r = 0; r < 8; ++r) {
    float inv = lrow[r] > 0.f ? 1.f / lrow[r] : 0.f;
    int ig = iw + r + 8 * hi;
#pragma unroll
    for (int c = 0; c < 4; ++c)
      CV[(size_t)(b * QLEN + ig) * DMODEL + h * DK + c * 16 + nl] =
          (_Float16)(oacc[c][r] * inv);
  }
}

// ---------------------------------------------------------------------------
extern "C" void kernel_launch(void* const* d_in, const int* in_sizes, int n_in,
                              void* d_out, int out_size, void* d_ws, size_t ws_size,
                              hipStream_t stream) {
  (void)in_sizes; (void)n_in; (void)out_size; (void)ws_size;
  // setup_inputs order: query(unused by reference!), key, pos_emb, mask,
  // u_bias, v_bias, Wk, Wv, Wq, Wp, Wo
  const float* key     = (const float*)d_in[1];
  const float* pos_emb = (const float*)d_in[2];
  const int*   mask    = (const int*)d_in[3];
  const float* u_bias  = (const float*)d_in[4];  // [H,dk] flat == per-column-n bias
  const float* v_bias  = (const float*)d_in[5];
  const float* Wk = (const float*)d_in[6];
  const float* Wv = (const float*)d_in[7];
  const float* Wq = (const float*)d_in[8];
  const float* Wp = (const float*)d_in[9];
  const float* Wo = (const float*)d_in[10];
  float* out = (float*)d_out;

  char* ws = (char*)d_ws;
  _Float16* K16  = (_Float16*)ws; ws += (size_t)BATCH * KLEN * DMODEL * 2;   // 4 MiB
  _Float16* V16  = (_Float16*)ws; ws += (size_t)BATCH * KLEN * DMODEL * 2;   // 4 MiB
  _Float16* QU16 = (_Float16*)ws; ws += (size_t)BATCH * QLEN * DMODEL * 2;   // 2 MiB
  _Float16* QV16 = (_Float16*)ws; ws += (size_t)BATCH * QLEN * DMODEL * 2;   // 2 MiB
  _Float16* P16  = (_Float16*)ws; ws += (size_t)KLEN * DMODEL * 2;           // 2 MiB
  _Float16* CV16 = (_Float16*)ws; ws += (size_t)BATCH * QLEN * DMODEL * 2;   // 2 MiB
  _Float16* SBD  = (_Float16*)ws;                                            // 64 MiB

  const dim3 blk(256);

  // Phase 1: projections (f32 in -> f16 out), bias fused into the Q copies.
  gemm_wt_kernel<float, _Float16><<<dim3(DMODEL / 64, BATCH * KLEN / 64), blk, 0, stream>>>(
      key, DMODEL, 11, KLEN, Wk, DMODEL, nullptr, K16, DMODEL, DMODEL);
  gemm_wt_kernel<float, _Float16><<<dim3(DMODEL / 64, BATCH * KLEN / 64), blk, 0, stream>>>(
      key, DMODEL, 11, KLEN, Wv, DMODEL, nullptr, V16, DMODEL, DMODEL);
  gemm_wt_kernel<float, _Float16><<<dim3(DMODEL / 64, BATCH * QLEN / 64), blk, 0, stream>>>(
      key, DMODEL, 10, KLEN, Wq, DMODEL, u_bias, QU16, DMODEL, DMODEL);  // q+u from key[:,-qlen:]
  gemm_wt_kernel<float, _Float16><<<dim3(DMODEL / 64, BATCH * QLEN / 64), blk, 0, stream>>>(
      key, DMODEL, 10, KLEN, Wq, DMODEL, v_bias, QV16, DMODEL, DMODEL);  // q+v
  gemm_wt_kernel<float, _Float16><<<dim3(DMODEL / 64, KLEN / 64), blk, 0, stream>>>(
      pos_emb, DMODEL, 11, KLEN, Wp, DMODEL, nullptr, P16, DMODEL, DMODEL);

  // Phase 2: pre-shift BD scores (L2-resident 64 MiB f16 tensor).
  bd_scores_kernel<<<dim3(KLEN / 64, QLEN / 64, BATCH * NHEAD), blk, 0, stream>>>(
      QV16, P16, SBD);

  // Phase 3: fused attention (AC + shifted BD + softmax + P~.V).
  attn_kernel<<<dim3(QLEN / 128, BATCH * NHEAD), blk, 0, stream>>>(
      QU16, K16, V16, SBD, mask, CV16);

  // Phase 4: output projection (f16 A, f32 out).
  gemm_wt_kernel<_Float16, float><<<dim3(DMODEL / 64, BATCH * QLEN / 64), blk, 0, stream>>>(
      CV16, DMODEL, 11, BATCH * QLEN, Wo, DMODEL, nullptr, out, DMODEL, DMODEL);
}
